// DualPathModule_43035572306593
// MI455X (gfx1250) — matrix operand order repaired and checked by reading, not compiled, not run on hardware
//
#include <hip/hip_runtime.h>
#include <hip/hip_bf16.h>

#define DEVINL __device__ __forceinline__

typedef __bf16 bf16_t;
typedef __attribute__((ext_vector_type(16))) __bf16 v16bf;
typedef __attribute__((ext_vector_type(8)))  float  v8f;

namespace {
constexpr int HID     = 256;
constexpr int NLAY    = 2;
constexpr int DSTATE  = 64;
constexpr int DCONV   = 4;
constexpr int DIN     = 512;
constexpr int NHEADS  = 8;
constexpr int HEADP   = 64;
constexpr int CONVDIM = 640;                  // DIN + 2*DSTATE
constexpr int PROJDIM = 1160;                 // 2*DIN + 2*DSTATE + NHEADS
constexpr int BB = 2, TT = 192, NBB = 62;
constexpr int NTOK  = BB * TT * NBB;          // 23808 (multiple of 16)
constexpr int NTOKP = 24576;                  // max padded tokens (384 * 64)
constexpr int NTIN  = 73;                     // ceil(1160/16) N-tiles, in-proj
constexpr int KTIN  = HID / 32;               // 8 K-steps, in-proj
constexpr int NTOUT = HID / 16;               // 16 N-tiles, out-proj
constexpr int KTOUT = DIN / 32;               // 16 K-steps, out-proj
}

// ---------------------------------------------------------------------------
// helpers
// ---------------------------------------------------------------------------
DEVINL long xoff(int path, int seq, int pos) {
  if (path == 0) {                       // t-path: seq = b*NB + nb, pos = t
    int b = seq / NBB, nb = seq - b * NBB;
    return (long)(b * TT + pos) * NBB + nb;
  }
  return (long)seq * NBB + pos;          // b-path: contiguous
}

DEVINL v8f zero_v8f() {
  v8f z = {0.f, 0.f, 0.f, 0.f, 0.f, 0.f, 0.f, 0.f};
  return z;
}

// A-operand fragment (16x32 bf16, MxK) from row-major memory (contiguous k order):
// lane m = lane&15; element j -> k = j + 16*lane[4]
// -> one contiguous 32B chunk per lane
DEVINL v16bf load_a_frag(const bf16_t* p, int ld) {
  int lane = threadIdx.x & 31;
  int m = lane & 15, hi = lane >> 4;
  v16bf a;
#pragma unroll
  for (int j = 0; j < 16; ++j) {
    int k = j + (hi << 4);
    a[j] = p[m * ld + k];
  }
  return a;
}

DEVINL float block256_reduce(float v) {
  __shared__ float red[8];
#pragma unroll
  for (int o = 16; o > 0; o >>= 1) v += __shfl_xor(v, o, 32);
  int w = threadIdx.x >> 5;
  if ((threadIdx.x & 31) == 0) red[w] = v;
  __syncthreads();
  float s = 0.f;
#pragma unroll
  for (int i = 0; i < 8; ++i) s += red[i];
  __syncthreads();
  return s;
}

DEVINL float siluf(float x) { return x / (1.f + __expf(-x)); }

// packed xh index: [seq][chunk][head][pfrag][lane][j]
DEVINL long xhp_index(int seq, int t, int c, int Lpad) {
  int nch = Lpad >> 5;
  int ch = t >> 5, sl = t & 31;
  int hi = sl >> 4, j = sl & 15;
  int head = c >> 6, pl = c & 63;
  int pf = pl >> 4, nl = pl & 15;
  int lane = nl + (hi << 4);
  return (((((long)seq * nch + ch) * NHEADS + head) * 4 + pf) * 32 + lane) * 16 + j;
}

// ---------------------------------------------------------------------------
// utility kernels
// ---------------------------------------------------------------------------
__global__ void k_copy(const float* __restrict__ s, float* __restrict__ d, int n) {
  int i = blockIdx.x * 256 + threadIdx.x;
  if (i < n) d[i] = s[i];
}

// pack weight [K,N] f32 -> WMMA B-fragment layout [kt][nt][lane][16] bf16,
// zero-padded in N.  element j of lane -> k = kt*32 + j + 16*lane[4],
// n = nt*16 + (lane&15)
__global__ void k_packw(const float* __restrict__ W, bf16_t* __restrict__ dst,
                        int N, int NT, int total) {
  int i = blockIdx.x * 256 + threadIdx.x;
  if (i >= total) return;
  int j = i & 15;
  int lane = (i >> 4) & 31;
  int tmp = i >> 9;
  int nt = tmp % NT, kt = tmp / NT;
  int k = kt * 32 + j + ((lane >> 4) << 4);
  int n = nt * 16 + (lane & 15);
  dst[i] = (n < N) ? (bf16_t)W[(long)k * N + n] : (bf16_t)0.0f;
}

// ---------------------------------------------------------------------------
// RMSNorm -> bf16 tokens [tok, HID]
// ---------------------------------------------------------------------------
__global__ void k_rmsnorm(const float* __restrict__ xw, const float* __restrict__ w,
                          bf16_t* __restrict__ out, int path, int Lp) {
  int tok = blockIdx.x;
  int seq = tok / Lp, pos = tok - seq * Lp;
  const float* xp = xw + xoff(path, seq, pos) * HID;
  int h = threadIdx.x;
  float v = xp[h];
  float s = block256_reduce(v * v);
  float rs = rsqrtf(s * (1.0f / HID) + 1e-6f);
  out[(long)tok * HID + h] = (bf16_t)(v * rs * w[h]);
}

// ---------------------------------------------------------------------------
// GEMM-in: [NTOK,256]bf16 x packed W -> [NTOK,1160]f32   (WMMA)
// ---------------------------------------------------------------------------
__global__ void k_gemm_in(const bf16_t* __restrict__ A, const bf16_t* __restrict__ WP,
                          float* __restrict__ C) {
  __shared__ __align__(16) bf16_t As[16 * HID];
  int mt = blockIdx.y * 16;
  {
    const uint4* src = (const uint4*)(A + (long)mt * HID);
    uint4* dst = (uint4*)As;
    dst[threadIdx.x]       = src[threadIdx.x];
    dst[threadIdx.x + 256] = src[threadIdx.x + 256];
  }
  __syncthreads();
  int wave = threadIdx.x >> 5;
  int ntile = blockIdx.x * 8 + wave;
  if (ntile >= NTIN) return;               // wave-uniform
  int lane = threadIdx.x & 31, col = lane & 15, hi = lane >> 4;
  v8f acc = zero_v8f();
#pragma unroll
  for (int kt = 0; kt < KTIN; ++kt) {
    v16bf a = load_a_frag(As + kt * 32, HID);
    v16bf b = *(const v16bf*)(WP + (((long)kt * NTIN + ntile) * 32 + lane) * 16);
    acc = __builtin_amdgcn_wmma_f32_16x16x32_bf16(false, a, false, b, (short)0, acc, false, false);
  }
  int n = ntile * 16 + col;
  if (n < PROJDIM) {
#pragma unroll
    for (int v = 0; v < 8; ++v)
      C[(long)(mt + v + hi * 8) * PROJDIM + n] = acc[v];
  }
}

// ---------------------------------------------------------------------------
// causal depthwise conv (K=4) + SiLU; outputs padded per sequence to Lpad
// ---------------------------------------------------------------------------
__global__ void k_conv(const float* __restrict__ zx, const float* __restrict__ wconv,
                       const float* __restrict__ bconv, float* __restrict__ xh,
                       bf16_t* __restrict__ xhp, bf16_t* __restrict__ bmb,
                       bf16_t* __restrict__ cmb, int Lp, int Lpad) {
  int seq = blockIdx.x;
  int idx = blockIdx.y * 256 + threadIdx.x;
  if (idx >= Lpad * CONVDIM) return;
  int t = idx / CONVDIM, c = idx - t * CONVDIM;
  long ptok = (long)seq * Lpad + t;
  if (t >= Lp) {                           // zero the pads
    if (c < DIN)              xhp[xhp_index(seq, t, c, Lpad)] = (bf16_t)0.0f;
    else if (c < DIN + DSTATE) bmb[ptok * DSTATE + (c - DIN)] = (bf16_t)0.0f;
    else                       cmb[ptok * DSTATE + (c - DIN - DSTATE)] = (bf16_t)0.0f;
    return;
  }
  long rowbase = ((long)seq * Lp) * PROJDIM + DIN + c;
  float acc = bconv[c];
#pragma unroll
  for (int k = 0; k < DCONV; ++k) {
    int tt = t - (DCONV - 1) + k;
    if (tt >= 0) acc += wconv[c * DCONV + k] * zx[rowbase + (long)tt * PROJDIM];
  }
  float s = siluf(acc);
  long tok = (long)seq * Lp + t;
  if (c < DIN) {
    xh[tok * DIN + c] = s;
    xhp[xhp_index(seq, t, c, Lpad)] = (bf16_t)s;
  } else if (c < DIN + DSTATE) {
    bmb[ptok * DSTATE + (c - DIN)] = (bf16_t)s;
  } else {
    cmb[ptok * DSTATE + (c - DIN - DSTATE)] = (bf16_t)s;
  }
}

// ---------------------------------------------------------------------------
// dt = softplus(raw + bias); fwd/bwd inclusive cumsums of dt*A; zero pads
// ---------------------------------------------------------------------------
__global__ void k_dtcum(const float* __restrict__ zx, const float* __restrict__ dt_bias,
                        const float* __restrict__ a_log, float* __restrict__ dt,
                        float* __restrict__ cumf, float* __restrict__ cumb,
                        int NSEQ, int Lp, int Lpad) {
  int i = blockIdx.x * 256 + threadIdx.x;
  if (i >= NSEQ * NHEADS) return;
  int seq = i / NHEADS, h = i - seq * NHEADS;
  float A = -__expf(a_log[h]);
  long tb  = (long)seq * Lp;               // real tokens (zx)
  long tbp = (long)seq * Lpad;             // padded
  float run = 0.f;
  for (int t = 0; t < Lp; ++t) {
    float v = zx[(tb + t) * PROJDIM + (PROJDIM - NHEADS) + h] + dt_bias[h];
    float d = (v > 20.f) ? v : log1pf(__expf(v));
    dt[(tbp + t) * NHEADS + h] = d;
    run += d * A;
    cumf[(tbp + t) * NHEADS + h] = run;
  }
  run = 0.f;
  for (int t = Lp - 1; t >= 0; --t) {
    run += dt[(tbp + t) * NHEADS + h] * A;
    cumb[(tbp + t) * NHEADS + h] = run;
  }
  for (int t = Lp; t < Lpad; ++t) {
    dt[(tbp + t) * NHEADS + h] = 0.f;
    cumf[(tbp + t) * NHEADS + h] = 0.f;
    cumb[(tbp + t) * NHEADS + h] = 0.f;
  }
}

// ---------------------------------------------------------------------------
// SSD (both directions fused): one wave per (seq, head, 16-row l-tile)
// ---------------------------------------------------------------------------
__global__ void k_ssd(const bf16_t* __restrict__ bmb, const bf16_t* __restrict__ cmb,
                      const bf16_t* __restrict__ xhp, const float* __restrict__ dt,
                      const float* __restrict__ cumf, const float* __restrict__ cumb,
                      float* __restrict__ y, int Lp, int Lpad) {
  __shared__ __align__(16) bf16_t sw[16 * 40];   // 16x32 score tile (padded stride)
  int lt = blockIdx.x, head = blockIdx.y, seq = blockIdx.z;
  int nch = Lpad >> 5;
  long tbp = (long)seq * Lpad;
  long tbr = (long)seq * Lp;
  int lane = threadIdx.x & 31;
  int col = lane & 15, hi = lane >> 4;

  float cl[8], sl[8];
#pragma unroll
  for (int v = 0; v < 8; ++v) {
    int l = lt * 16 + v + hi * 8;                // < Lpad always
    cl[v] = cumf[(tbp + l) * NHEADS + head];
    sl[v] = cumb[(tbp + l) * NHEADS + head];
  }

  v8f acc[4];
#pragma unroll
  for (int i = 0; i < 4; ++i) acc[i] = zero_v8f();

  for (int sc = 0; sc < nch; ++sc) {
#pragma unroll
    for (int sub = 0; sub < 2; ++sub) {
      int s0 = sc * 32 + sub * 16;
      v8f g = zero_v8f();
#pragma unroll
      for (int ks = 0; ks < DSTATE; ks += 32) {
        // A = C rows of the l-tile (16 contiguous bf16 per lane)
        v16bf a;
        {
          const bf16_t* cp = cmb + (tbp + lt * 16 + col) * DSTATE;
#pragma unroll
          for (int j = 0; j < 16; ++j) {
            int k = ks + j + (hi << 4);
            a[j] = cp[k];
          }
        }
        // B = B^T column s = B row s : 16 contiguous bf16 per lane
        v16bf b = *(const v16bf*)(bmb + (tbp + s0 + col) * DSTATE + ks + hi * 16);
        g = __builtin_amdgcn_wmma_f32_16x16x32_bf16(false, a, false, b, (short)0, g, false, false);
      }
      // decay/dt weighting on the D-fragment: element v -> (l, s)
      int s = s0 + col;
      float cs = cumf[(tbp + s) * NHEADS + head];
      float sb = cumb[(tbp + s) * NHEADS + head];
      float ds = dt[(tbp + s) * NHEADS + head];
      bool sok = s < Lp;
#pragma unroll
      for (int v = 0; v < 8; ++v) {
        int m = v + hi * 8;
        int l = lt * 16 + m;
        float w = 0.f;
        if (sok && l < Lp) {
          if (s <= l) w += __expf(cl[v] - cs);   // forward  (exponent <= 0)
          if (s >= l) w += __expf(sl[v] - sb);   // backward (exponent <= 0)
          w *= ds;
        }
        sw[m * 40 + sub * 16 + col] = (bf16_t)(g[v] * w);
      }
    }
    __syncthreads();
    // y(16x64) += S(16x32) . X(32x64); X loaded as packed fragments
    v16bf a2 = load_a_frag(sw, 40);
#pragma unroll
    for (int pf = 0; pf < 4; ++pf) {
      v16bf b = *(const v16bf*)(
          xhp + (((((long)seq * nch + sc) * NHEADS + head) * 4 + pf) * 32 + lane) * 16);
      acc[pf] = __builtin_amdgcn_wmma_f32_16x16x32_bf16(false, a2, false, b, (short)0, acc[pf], false, false);
    }
    __syncthreads();
  }
#pragma unroll
  for (int pf = 0; pf < 4; ++pf) {
#pragma unroll
    for (int v = 0; v < 8; ++v) {
      int l = lt * 16 + v + hi * 8;
      if (l < Lp) y[(tbr + l) * DIN + head * HEADP + pf * 16 + col] = acc[pf][v];
    }
  }
}

// ---------------------------------------------------------------------------
// gate: y += D*xh;  g = y*silu(z);  out = rmsnorm(g)*gn_w  -> bf16
// ---------------------------------------------------------------------------
__global__ void k_gate(const float* __restrict__ y, const float* __restrict__ xh,
                       const float* __restrict__ zx, const float* __restrict__ Dp,
                       const float* __restrict__ gnw, bf16_t* __restrict__ out) {
  long tok = blockIdx.x;
  float g[2];
  float ss = 0.f;
#pragma unroll
  for (int r = 0; r < 2; ++r) {
    int c = threadIdx.x + r * 256;
    float t = y[tok * DIN + c] + Dp[c >> 6] * xh[tok * DIN + c];
    float z = zx[tok * PROJDIM + c];
    g[r] = t * siluf(z);
    ss += g[r] * g[r];
  }
  float tot = block256_reduce(ss);
  float rs = rsqrtf(tot * (1.0f / DIN) + 1e-6f);
#pragma unroll
  for (int r = 0; r < 2; ++r) {
    int c = threadIdx.x + r * 256;
    out[tok * DIN + c] = (bf16_t)(g[r] * rs * gnw[c]);
  }
}

// ---------------------------------------------------------------------------
// GEMM-out: [NTOK,512]bf16 x packed W, residual-add into working x (WMMA)
// ---------------------------------------------------------------------------
__global__ void k_gemm_out(const bf16_t* __restrict__ A, const bf16_t* __restrict__ WP,
                           float* __restrict__ xw, int path, int Lp) {
  __shared__ __align__(16) bf16_t As[16 * DIN];
  int mt = blockIdx.y * 16;
  {
    const uint4* src = (const uint4*)(A + (long)mt * DIN);
    uint4* dst = (uint4*)As;
#pragma unroll
    for (int r = 0; r < 4; ++r)
      dst[threadIdx.x + r * 256] = src[threadIdx.x + r * 256];
  }
  __syncthreads();
  int wave = threadIdx.x >> 5;
  int ntile = blockIdx.x * 8 + wave;           // < 16 always
  int lane = threadIdx.x & 31, col = lane & 15, hi = lane >> 4;
  v8f acc = zero_v8f();
#pragma unroll
  for (int kt = 0; kt < KTOUT; ++kt) {
    v16bf a = load_a_frag(As + kt * 32, DIN);
    v16bf b = *(const v16bf*)(WP + (((long)kt * NTOUT + ntile) * 32 + lane) * 16);
    acc = __builtin_amdgcn_wmma_f32_16x16x32_bf16(false, a, false, b, (short)0, acc, false, false);
  }
  int n = ntile * 16 + col;
#pragma unroll
  for (int v = 0; v < 8; ++v) {
    int m = mt + v + hi * 8;
    int seq = m / Lp, pos = m - seq * Lp;
    long off = xoff(path, seq, pos) * HID + n;
    xw[off] += acc[v];                         // unique (m,n) per launch
  }
}

// ---------------------------------------------------------------------------
// host orchestration
// ---------------------------------------------------------------------------
extern "C" void kernel_launch(void* const* d_in, const int* in_sizes, int n_in,
                              void* d_out, int out_size, void* d_ws, size_t ws_size,
                              hipStream_t stream) {
  (void)in_sizes; (void)n_in; (void)out_size; (void)ws_size;

  size_t off = 0;
  auto take = [&](size_t bytes) -> char* {
    char* p = (char*)d_ws + off;
    off += (bytes + 255) & ~(size_t)255;
    return p;
  };
  float*  XW   = (float*) take((size_t)NTOK * HID * 4);
  bf16_t* XN   = (bf16_t*)take((size_t)NTOK * HID * 2);
  float*  ZX   = (float*) take((size_t)NTOK * PROJDIM * 4);
  float*  XH   = (float*) take((size_t)NTOK * DIN * 4);
  bf16_t* XHP  = (bf16_t*)take((size_t)NTOKP * DIN * 2);
  bf16_t* BMB  = (bf16_t*)take((size_t)NTOKP * DSTATE * 2);
  bf16_t* CMB  = (bf16_t*)take((size_t)NTOKP * DSTATE * 2);
  float*  DTB  = (float*) take((size_t)NTOKP * NHEADS * 4);
  float*  CUF  = (float*) take((size_t)NTOKP * NHEADS * 4);
  float*  CUB  = (float*) take((size_t)NTOKP * NHEADS * 4);
  float*  Y    = (float*) take((size_t)NTOK * DIN * 4);
  bf16_t* YNB  = (bf16_t*)take((size_t)NTOK * DIN * 2);
  const size_t WINSZ = (size_t)KTIN * NTIN * 512;    // per (path,layer)
  const size_t WOSZ  = (size_t)KTOUT * NTOUT * 512;
  bf16_t* WPIN = (bf16_t*)take(4 * WINSZ * 2);
  bf16_t* WPO  = (bf16_t*)take(4 * WOSZ * 2);

  // working copy of x (never mutate d_in)
  k_copy<<<dim3((NTOK * HID + 255) / 256), 256, 0, stream>>>(
      (const float*)d_in[0], XW, NTOK * HID);

  // pack projection weights into WMMA B-fragment layout
  for (int p = 0; p < 2; ++p) {
    const float* win  = (const float*)d_in[p ? 11 : 2];
    const float* wout = (const float*)d_in[p ? 18 : 9];
    for (int l = 0; l < NLAY; ++l) {
      int wi = p * NLAY + l;
      k_packw<<<dim3(((int)WINSZ + 255) / 256), 256, 0, stream>>>(
          win + (size_t)l * HID * PROJDIM, WPIN + (size_t)wi * WINSZ,
          PROJDIM, NTIN, (int)WINSZ);
      k_packw<<<dim3(((int)WOSZ + 255) / 256), 256, 0, stream>>>(
          wout + (size_t)l * DIN * HID, WPO + (size_t)wi * WOSZ,
          HID, NTOUT, (int)WOSZ);
    }
  }

  for (int l = 0; l < NLAY; ++l) {
    for (int p = 0; p < 2; ++p) {
      int di = p ? 9 : 0;
      const float* norm_w  = (const float*)d_in[1 + di] + (size_t)l * HID;
      const float* wconv   = (const float*)d_in[3 + di] + (size_t)l * CONVDIM * DCONV;
      const float* bconv   = (const float*)d_in[4 + di] + (size_t)l * CONVDIM;
      const float* dt_bias = (const float*)d_in[5 + di] + (size_t)l * NHEADS;
      const float* a_log   = (const float*)d_in[6 + di] + (size_t)l * NHEADS;
      const float* Dv      = (const float*)d_in[7 + di] + (size_t)l * NHEADS;
      const float* gn_w    = (const float*)d_in[8 + di] + (size_t)l * DIN;
      int NSEQ = p ? BB * TT : BB * NBB;        // 384 or 124
      int Lp   = p ? NBB : TT;                  // 62  or 192
      int Lpad = p ? 64 : 192;                  // per-seq padded length
      int LT   = Lpad / 16;
      int wi   = p * NLAY + l;

      k_rmsnorm<<<dim3(NTOK), 256, 0, stream>>>(XW, norm_w, XN, p, Lp);
      k_gemm_in<<<dim3((NTIN + 7) / 8, NTOK / 16), 256, 0, stream>>>(
          XN, WPIN + (size_t)wi * WINSZ, ZX);
      k_conv<<<dim3(NSEQ, (Lpad * CONVDIM + 255) / 256), 256, 0, stream>>>(
          ZX, wconv, bconv, XH, XHP, BMB, CMB, Lp, Lpad);
      k_dtcum<<<dim3((NSEQ * NHEADS + 255) / 256), 256, 0, stream>>>(
          ZX, dt_bias, a_log, DTB, CUF, CUB, NSEQ, Lp, Lpad);
      k_ssd<<<dim3(LT, NHEADS, NSEQ), 32, 0, stream>>>(
          BMB, CMB, XHP, DTB, CUF, CUB, Y, Lp, Lpad);
      k_gate<<<dim3(NTOK), 256, 0, stream>>>(Y, XH, ZX, Dv, gn_w, YNB);
      k_gemm_out<<<dim3(2, NTOK / 16), 256, 0, stream>>>(
          YNB, WPO + (size_t)wi * WOSZ, XW, p, Lp);
    }
  }

  k_copy<<<dim3((NTOK * HID + 255) / 256), 256, 0, stream>>>(XW, (float*)d_out, NTOK * HID);
}
